// Loss_82025285419754
// MI455X (gfx1250) — matrix operand order repaired and checked by reading, not compiled, not run on hardware
//
#include <hip/hip_runtime.h>
#include <hip/hip_bf16.h>
#include <math.h>

// ---------------------------------------------------------------------------
// NT-Xent contrastive loss, fused GEMM(+WMMA bf16x3) + online logsumexp.
//   B=4096, D=256, N=8192, temp=0.5 (i.e. scale sim by 2.0)
// Never materializes the 8192x8192 sim matrix. Inputs (8MB) are L2-resident.
// bf16 hi/lo split recovers ~fp32 accuracy: sim ~= hi.hi^T + hi.lo^T + lo.hi^T
// B-tiles are staged LDS-double-buffered via GLOBAL_LOAD_ASYNC_TO_LDS_B128
// (ASYNCcnt) so the L2 latency is hidden behind the WMMA block.
// ---------------------------------------------------------------------------

#define BB   4096
#define DD   256
#define NN   8192
#define NEGBIG (-3.0e38f)

typedef __attribute__((ext_vector_type(16))) __bf16 v16bf;
typedef __attribute__((ext_vector_type(8)))  float  v8f;

union Frag {
    uint4 q[2];   // 32 bytes
    v16bf v;      // 16 bf16 elements
};

// async copy 16B global -> LDS, tracked by ASYNCcnt (CDNA5)
__device__ __forceinline__ void async_ld16(const __bf16* g, __bf16* l) {
    const unsigned lds_off = (unsigned)(uintptr_t)l;   // aperture low 32b = LDS offset
    asm volatile("global_load_async_to_lds_b128 %0, %1, off"
                 :: "v"(lds_off), "v"(g)
                 : "memory");
}
__device__ __forceinline__ void wait_async0() {
    asm volatile("s_wait_asynccnt 0x0" ::: "memory");
}

// ---- kernel 1: split concat(h_i,h_j) into bf16 hi + bf16 lo arrays --------
__global__ __launch_bounds__(256) void ntxent_prep(
    const float* __restrict__ h_i, const float* __restrict__ h_j,
    __bf16* __restrict__ hi, __bf16* __restrict__ lo)
{
    const int idx = blockIdx.x * 256 + threadIdx.x;
    if (idx >= NN * DD) return;
    const int row = idx >> 8;         // / DD
    const int d   = idx & (DD - 1);
    const float x = (row < BB) ? h_i[row * DD + d] : h_j[(row - BB) * DD + d];
    const __bf16 h = (__bf16)x;
    hi[idx] = h;
    lo[idx] = (__bf16)(x - (float)h);
}

// ---- kernel 2: fused GEMM + masked online logsumexp -----------------------
// 64 blocks x 256 threads (8 waves). Each wave owns a 16-row strip; the block
// stages each 16-col B-tile (hi+lo, K=256) in a double-buffered LDS pipeline.
__global__ __launch_bounds__(256) void ntxent_main(
    const __bf16* __restrict__ hi, const __bf16* __restrict__ lo,
    float* __restrict__ partial)
{
    // padded row stride 264 (=33 uint4) -> LDS bank stride 4, conflict-free
    __shared__ __align__(16) __bf16 Bhi[2][16][264];
    __shared__ __align__(16) __bf16 Blo[2][16][264];
    __shared__ float mred[8][16][16];
    __shared__ float sred[8][16][16];
    __shared__ float lsered[8][16];
    __shared__ float posred[8][32];

    const int lane    = threadIdx.x & 31;
    const int wave    = threadIdx.x >> 5;
    const int halfsel = (lane < 16) ? 0 : 1;
    const int ncol_l  = lane & 15;
    const int m0      = (blockIdx.x * 8 + wave) * 16;
    const int mrow    = m0 + ncol_l;             // A-frag row for this lane

    // staging assignment for this thread: 2 x (hi+lo) 16B chunks per tile
    const int sr0 = threadIdx.x >> 5;            // row 0..7
    const int sc0 = threadIdx.x & 31;            // uint4 col 0..31
    const int sr1 = sr0 + 8;                     // row 8..15

    // Pin A fragments (this wave's 16 rows, full K=256, hi+lo) in VGPRs.
    // A 16x32 bf16 layout: lane<16 -> K {0..7, 16..23}; lane>=16 -> +8.
    Frag ahi[8], alo[8];
#pragma unroll
    for (int kc = 0; kc < 8; ++kc) {
        const int kb = kc * 32 + halfsel * 8;
        const uint4* ph = (const uint4*)(hi + (size_t)mrow * DD + kb);
        const uint4* pl = (const uint4*)(lo + (size_t)mrow * DD + kb);
        ahi[kc].q[0] = ph[0]; ahi[kc].q[1] = ph[2];   // +16 elems = +2 uint4
        alo[kc].q[0] = pl[0]; alo[kc].q[1] = pl[2];
    }

    float m[8], s[8];
    float posacc = 0.0f;
#pragma unroll
    for (int v = 0; v < 8; ++v) { m[v] = NEGBIG; s[v] = 0.0f; }

    // prologue: stage tile 0 into buffer 0 (async -> wait -> barrier)
    {
        const size_t g0 = (size_t)sr0 * DD + sc0 * 8;
        const size_t g1 = (size_t)sr1 * DD + sc0 * 8;
        async_ld16(hi + g0, &Bhi[0][sr0][sc0 * 8]);
        async_ld16(lo + g0, &Blo[0][sr0][sc0 * 8]);
        async_ld16(hi + g1, &Bhi[0][sr1][sc0 * 8]);
        async_ld16(lo + g1, &Blo[0][sr1][sc0 * 8]);
    }
    wait_async0();
    __syncthreads();

    for (int nt = 0; nt < NN / 16; ++nt) {
        const int p = nt & 1;

        // kick off async stage of the NEXT tile into the other buffer
        if (nt + 1 < NN / 16) {
            const int n1 = (nt + 1) * 16;
            const size_t g0 = (size_t)(n1 + sr0) * DD + sc0 * 8;
            const size_t g1 = (size_t)(n1 + sr1) * DD + sc0 * 8;
            async_ld16(hi + g0, &Bhi[1 - p][sr0][sc0 * 8]);
            async_ld16(lo + g0, &Blo[1 - p][sr0][sc0 * 8]);
            async_ld16(hi + g1, &Bhi[1 - p][sr1][sc0 * 8]);
            async_ld16(lo + g1, &Blo[1 - p][sr1][sc0 * 8]);
        }

        v8f c = {0.f, 0.f, 0.f, 0.f, 0.f, 0.f, 0.f, 0.f};
#pragma unroll
        for (int kc = 0; kc < 8; ++kc) {
            // B 32x16 bf16 layout: lane<16 -> K {0..7, 16..23}; lane>=16 -> +8
            const int kb = kc * 32 + halfsel * 8;
            Frag bh, bl;
            const uint4* pbh = (const uint4*)&Bhi[p][ncol_l][kb];
            const uint4* pbl = (const uint4*)&Blo[p][ncol_l][kb];
            bh.q[0] = pbh[0]; bh.q[1] = pbh[2];
            bl.q[0] = pbl[0]; bl.q[1] = pbl[2];
            c = __builtin_amdgcn_wmma_f32_16x16x32_bf16(false, ahi[kc].v, false, bh.v,
                                                        (short)0, c, false, false);
            c = __builtin_amdgcn_wmma_f32_16x16x32_bf16(false, ahi[kc].v, false, bl.v,
                                                        (short)0, c, false, false);
            c = __builtin_amdgcn_wmma_f32_16x16x32_bf16(false, alo[kc].v, false, bh.v,
                                                        (short)0, c, false, false);
        }

        // epilogue on C layout: elem v, lane l -> row m0+v+(l<16?0:8), col n0+(l&15)
        const int col = nt * 16 + ncol_l;
#pragma unroll
        for (int v = 0; v < 8; ++v) {
            const int row = m0 + v + halfsel * 8;
            float x = c[v] * 2.0f;                       // / TEMP_F (0.5)
            x = (col == row) ? NEGBIG : x;               // mask self-similarity
            const int poscol = (row < BB) ? (row + BB) : (row - BB);
            if (col == poscol) posacc += x;              // positive-pair logit
            // one-exp online logsumexp update
            const float mold = m[v];
            const float e = __expf(-fabsf(x - mold));
            const bool up = x > mold;
            s[v] = up ? (s[v] * e + 1.0f) : (s[v] + e);
            m[v] = up ? x : mold;
        }

        // pipeline handoff: my async writes done + everyone's reads done
        wait_async0();
        __syncthreads();
    }

    // cross-lane combine: row r stats live in 16 lanes of one half-wave
#pragma unroll
    for (int v = 0; v < 8; ++v) {
        const int rl = v + halfsel * 8;
        mred[wave][rl][ncol_l] = m[v];
        sred[wave][rl][ncol_l] = s[v];
    }
    posred[wave][lane] = posacc;
    __syncthreads();

    if (lane < 16) {
        float M = NEGBIG;
#pragma unroll
        for (int j = 0; j < 16; ++j) M = fmaxf(M, mred[wave][lane][j]);
        float S = 0.0f;
#pragma unroll
        for (int j = 0; j < 16; ++j) S += sred[wave][lane][j] * __expf(mred[wave][lane][j] - M);
        lsered[wave][lane] = __logf(S) + M;
    }
    __syncthreads();

    if (lane == 0) {
        float acc = 0.0f;
        for (int j = 0; j < 16; ++j) acc += lsered[wave][j];
        for (int j = 0; j < 32; ++j) acc -= posred[wave][j];
        partial[blockIdx.x * 8 + wave] = acc;            // sum(lse - pos) for 16 rows
    }
}

// ---- kernel 3: deterministic tree reduction of 512 partials ---------------
__global__ __launch_bounds__(256) void ntxent_reduce(
    const float* __restrict__ partial, float* __restrict__ out)
{
    __shared__ float red[256];
    const int t = threadIdx.x;
    red[t] = partial[t] + partial[t + 256];
    __syncthreads();
    for (int ofs = 128; ofs > 0; ofs >>= 1) {
        if (t < ofs) red[t] += red[t + ofs];
        __syncthreads();
    }
    if (t == 0) out[0] = red[0] * (1.0f / (float)NN);
}

// ---------------------------------------------------------------------------
extern "C" void kernel_launch(void* const* d_in, const int* in_sizes, int n_in,
                              void* d_out, int out_size, void* d_ws, size_t ws_size,
                              hipStream_t stream)
{
    (void)in_sizes; (void)n_in; (void)out_size; (void)ws_size;
    const float* h_i = (const float*)d_in[0];
    const float* h_j = (const float*)d_in[1];
    float* out = (float*)d_out;

    // workspace: [0,4MB) hi bf16, [4MB,8MB) lo bf16, [8MB,+2KB) partials
    char* ws = (char*)d_ws;
    __bf16* hi      = (__bf16*)(ws);
    __bf16* lo      = (__bf16*)(ws + (size_t)NN * DD * sizeof(__bf16));
    float*  partial = (float*)(ws + (size_t)NN * DD * sizeof(__bf16) * 2);

    ntxent_prep<<<(NN * DD + 255) / 256, 256, 0, stream>>>(h_i, h_j, hi, lo);
    ntxent_main<<<64, 256, 0, stream>>>(hi, lo, partial);
    ntxent_reduce<<<1, 256, 0, stream>>>(partial, out);
}
